// SelfAttentionModel_54004918780607
// MI455X (gfx1250) — compile-verified, matrix operand order checked by reading, not run on hardware
//
#include <hip/hip_runtime.h>
#include <hip/hip_bf16.h>

typedef __attribute__((ext_vector_type(16))) _Float16 v16h;
typedef __attribute__((ext_vector_type(8)))  _Float16 v8h;
typedef __attribute__((ext_vector_type(8)))  float    v8f;

constexpr int Bb = 4;      // batch
constexpr int Nn = 4096;   // H*W
constexpr int Cc = 256;    // channels
constexpr int Dd = 32;     // C/8 (q,k dim)
constexpr int SP = 4096 + 8;   // padded LDS row stride in halves (bank de-conflict)
constexpr int WAVES = 8;       // waves per attention workgroup

// Load a 16x32 f16 A/B WMMA operand row-slice: two contiguous 16B chunks
// at half-offsets kbase and kbase+16 (ISA 16-bit operand layout).
__device__ inline v16h load_op16(const _Float16* rowp, int kbase) {
    union { v16h v; v8h h[2]; } u;
    u.h[0] = *(const v8h*)(rowp + kbase);
    u.h[1] = *(const v8h*)(rowp + kbase + 16);
    return u.v;
}

__device__ inline v8f wmma_f16(v16h a, v16h b, v8f c) {
    return __builtin_amdgcn_wmma_f32_16x16x32_f16(
        false, a, false, b, (short)0, c, false, false);
}

// ---------------------------------------------------------------------------
// Kernel 1: generic 1x1-conv projection, launched separately for Q, K, V.
// One wave computes one 16x16 tile of (N x 256) @ (256 x 16*nct).
// Destination addressing is branch-free via row/col/batch strides:
//   Q,K row-major f16:  rowStride=Dd, colStride=1,  batchStride=N*Dd
//   V transposed (CxN): rowStride=1,  colStride=Nn, batchStride=C*Nn
// ---------------------------------------------------------------------------
__global__ __launch_bounds__(32) void proj_kernel(
    const float* __restrict__ x, const float* __restrict__ W,
    const float* __restrict__ bias, _Float16* __restrict__ outp,
    int Dw, int nct, long rowStride, long colStride, long batchStride)
{
    int blk  = blockIdx.x;
    int ct   = blk % nct;
    int mt   = (blk / nct) % (Nn/16);
    int b    = blk / (nct * (Nn/16));
    int lane = threadIdx.x;
    int lrow = lane & 15;
    int kbase = (lane < 16) ? 0 : 8;     // ISA 16-bit A/B operand K layout
    int rbase = (lane >= 16) ? 8 : 0;
    int m0   = mt * 16;
    int col  = ct * 16 + lrow;

    const float* xrow = x + ((size_t)(b * Nn + m0 + lrow)) * Cc;

    v8f acc = {};
    for (int kc = 0; kc < Cc; kc += 32) {
        // A from X: 16 fp32 -> f16, as 4 x float4 vector loads
        float4 f0 = *(const float4*)(xrow + kc + kbase);
        float4 f1 = *(const float4*)(xrow + kc + kbase + 4);
        float4 f2 = *(const float4*)(xrow + kc + kbase + 16);
        float4 f3 = *(const float4*)(xrow + kc + kbase + 20);
        v16h a;
        a[0]=(_Float16)f0.x;  a[1]=(_Float16)f0.y;  a[2]=(_Float16)f0.z;  a[3]=(_Float16)f0.w;
        a[4]=(_Float16)f1.x;  a[5]=(_Float16)f1.y;  a[6]=(_Float16)f1.z;  a[7]=(_Float16)f1.w;
        a[8]=(_Float16)f2.x;  a[9]=(_Float16)f2.y;  a[10]=(_Float16)f2.z; a[11]=(_Float16)f2.w;
        a[12]=(_Float16)f3.x; a[13]=(_Float16)f3.y; a[14]=(_Float16)f3.z; a[15]=(_Float16)f3.w;

        // B from W: gather down the k dimension (W is tiny and L2-hot)
        v16h bm;
        #pragma unroll
        for (int h = 0; h < 16; ++h) {
            int k = kc + kbase + (h < 8 ? h : h + 8);
            bm[h] = (_Float16)W[(size_t)k * Dw + col];
        }
        acc = wmma_f16(a, bm, acc);
    }

    float bias_c = bias[col];
    _Float16* op = outp + (size_t)b * batchStride
                        + (size_t)(m0 + rbase) * rowStride
                        + (size_t)col * colStride;
    #pragma unroll
    for (int r = 0; r < 8; ++r)
        op[(size_t)r * rowStride] = (_Float16)(acc[r] + bias_c);
}

// ---------------------------------------------------------------------------
// Kernel 2: attention.  One workgroup (8 waves) per (batch, 16-query-row
// tile).  Full 16 x 4096 score block held in LDS (f16, padded stride).
// ---------------------------------------------------------------------------
__global__ __launch_bounds__(WAVES * 32) void attn_kernel(
    const float* __restrict__ x,
    const _Float16* __restrict__ Qh, const _Float16* __restrict__ Kh,
    const _Float16* __restrict__ Vth,
    const float* __restrict__ gamma_p, float* __restrict__ out)
{
    extern __shared__ char lds_raw[];
    _Float16* Sh     = (_Float16*)lds_raw;                 // 16 * SP halves
    float*    red    = (float*)(lds_raw + 16 * SP * 2);    // 16 * 16
    float*    rowmax = red + 16 * 16;                      // 16
    float*    rowinv = rowmax + 16;                        // 16

    int b    = blockIdx.x >> 8;        // 256 row tiles per batch
    int mt   = blockIdx.x & 255;
    int m0   = mt * 16;
    int wave = threadIdx.x >> 5;
    int lane = threadIdx.x & 31;
    int lrow = lane & 15;
    int kbase = (lane < 16) ? 0 : 8;
    int rbase = (lane >= 16) ? 8 : 0;

    // Q tile as A operand (reused by all key tiles of this wave)
    v16h aq = load_op16(Qh + ((size_t)(b * Nn + m0 + lrow)) * Dd, kbase);

    // ---- Phase 1: S = Q K^T  ->  LDS (f16) -------------------------------
    for (int jt = wave; jt < Nn / 16; jt += WAVES) {
        int nk = jt * 16;
        v16h bk_ = load_op16(Kh + ((size_t)(b * Nn + nk + lrow)) * Dd, kbase);
        v8f c = {};
        c = wmma_f16(aq, bk_, c);
        #pragma unroll
        for (int r = 0; r < 8; ++r)
            Sh[(size_t)(r + rbase) * SP + nk + lrow] = (_Float16)c[r];
    }
    __syncthreads();

    // ---- Phase 2: softmax stats; store exp(s-max) in place ---------------
    {
        int row = threadIdx.x >> 4;            // 16 rows * 16 segments
        int seg = threadIdx.x & 15;
        const int SEG = Nn / 16;               // 256 elements per segment
        v8h* srowv = (v8h*)(Sh + (size_t)row * SP + seg * SEG);

        float mx = -3.0e38f;
        for (int i = 0; i < SEG / 8; ++i) {
            v8h c = srowv[i];
            #pragma unroll
            for (int j = 0; j < 8; ++j) mx = fmaxf(mx, (float)c[j]);
        }
        red[row * 16 + seg] = mx;
        __syncthreads();
        if (seg == 0) {
            float m = red[row * 16];
            for (int s = 1; s < 16; ++s) m = fmaxf(m, red[row * 16 + s]);
            rowmax[row] = m;
        }
        __syncthreads();
        float m = rowmax[row];
        float sum = 0.f;
        for (int i = 0; i < SEG / 8; ++i) {
            v8h c = srowv[i];
            #pragma unroll
            for (int j = 0; j < 8; ++j) {
                float e = __expf((float)c[j] - m);
                c[j] = (_Float16)e;            // exp(s-max) <= 1, f16-safe
                sum += e;
            }
            srowv[i] = c;
        }
        red[row * 16 + seg] = sum;
        __syncthreads();
        if (seg == 0) {
            float s = 0.f;
            for (int t = 0; t < 16; ++t) s += red[row * 16 + t];
            rowinv[row] = 1.0f / s;
        }
        __syncthreads();
    }

    // ---- Phase 3: O = P V with dual accumulators per wave ----------------
    // 16 column tiles / 8 waves = 2 tiles each; A operand loaded once per
    // k-chunk and shared by both B operands (halves LDS traffic, gives two
    // independent wmma chains to overlap with loads).
    float g = gamma_p[0];
    const _Float16* srow = Sh + (size_t)lrow * SP;
    {
        int n0 = (wave * 2) * 16;
        int n1 = n0 + 16;
        const _Float16* vrow0 = Vth + ((size_t)(b * Cc) + n0 + lrow) * Nn;
        const _Float16* vrow1 = vrow0 + (size_t)16 * Nn;
        v8f acc0 = {}, acc1 = {};
        for (int kc = 0; kc < Nn; kc += 32) {
            v16h a  = load_op16(srow  + kc, kbase);   // ds_load_b128 x2
            v16h b0 = load_op16(vrow0 + kc, kbase);   // global_load_b128 x2
            v16h b1 = load_op16(vrow1 + kc, kbase);
            acc0 = wmma_f16(a, b0, acc0);
            acc1 = wmma_f16(a, b1, acc1);
        }
        #pragma unroll
        for (int r = 0; r < 8; ++r) {
            int row = r + rbase;
            float scale = g * rowinv[row];
            size_t base = ((size_t)(b * Nn) + m0 + row) * Cc + lrow;
            size_t i0 = base + n0, i1 = base + n1;
            out[i0] = scale * acc0[r] + x[i0];
            out[i1] = scale * acc1[r] + x[i1];
        }
    }
}

// ---------------------------------------------------------------------------
extern "C" void kernel_launch(void* const* d_in, const int* in_sizes, int n_in,
                              void* d_out, int out_size, void* d_ws, size_t ws_size,
                              hipStream_t stream) {
    (void)in_sizes; (void)n_in; (void)out_size; (void)ws_size;
    const float* x     = (const float*)d_in[0];
    const float* Wq    = (const float*)d_in[1];
    const float* bq    = (const float*)d_in[2];
    const float* Wk    = (const float*)d_in[3];
    const float* bk    = (const float*)d_in[4];
    const float* Wv    = (const float*)d_in[5];
    const float* bv    = (const float*)d_in[6];
    const float* gamma = (const float*)d_in[7];
    float* out = (float*)d_out;

    _Float16* Qh  = (_Float16*)d_ws;                         // B*N*D
    _Float16* Kh  = Qh + (size_t)Bb * Nn * Dd;               // B*N*D
    _Float16* Vth = Kh + (size_t)Bb * Nn * Dd;               // B*C*N (transposed)

    // Q: 2 col tiles, row-major out
    proj_kernel<<<Bb * (Nn / 16) * 2, 32, 0, stream>>>(
        x, Wq, bq, Qh, Dd, 2, Dd, 1, (long)Nn * Dd);
    // K: 2 col tiles, row-major out
    proj_kernel<<<Bb * (Nn / 16) * 2, 32, 0, stream>>>(
        x, Wk, bk, Kh, Dd, 2, Dd, 1, (long)Nn * Dd);
    // V: 16 col tiles, transposed out (C x N)
    proj_kernel<<<Bb * (Nn / 16) * 16, 32, 0, stream>>>(
        x, Wv, bv, Vth, Cc, 16, 1, Nn, (long)Cc * Nn);

    size_t shmem = (size_t)16 * SP * 2 + (16 * 16 + 16 + 16) * sizeof(float);
    attn_kernel<<<Bb * (Nn / 16), WAVES * 32, shmem, stream>>>(
        x, Qh, Kh, Vth, gamma, out);
}
